// G2C_52055003627988
// MI455X (gfx1250) — matrix operand order repaired and checked by reading, not compiled, not run on hardware
//
#include <hip/hip_runtime.h>

#define BSZ   2048
#define NMAX  21
#define NN    441            // 21*21
#define ROWS  (BSZ * NN)     // 903168
#define KDIM  128
#define HDIM  128

typedef __attribute__((ext_vector_type(16))) __bf16 v16bf;
typedef __attribute__((ext_vector_type(8)))  float  v8f;

__device__ __forceinline__ unsigned short bf16rne(float f) {
    unsigned u = __float_as_uint(f);
    unsigned r = u + 0x7FFFu + ((u >> 16) & 1u);
    return (unsigned short)(r >> 16);
}

__device__ __forceinline__ unsigned pcg_hash(unsigned v) {
    v = v * 747796405u + 2891336453u;
    unsigned w = ((v >> ((v >> 28u) + 4u)) ^ v) * 277803737u;
    return (w >> 22u) ^ w;
}

__device__ __forceinline__ float gaussf(unsigned s) {
    unsigned a = pcg_hash(s);
    unsigned b = pcg_hash(s ^ 0x9E3779B9u);
    float u1 = (float)(a + 1u) * 2.3283064e-10f;   // (0,1]
    float u2 = (float)b * 2.3283064e-10f;
    return sqrtf(-2.0f * logf(fmaxf(u1, 1e-12f))) * cosf(6.2831853f * u2);
}

__device__ __forceinline__ float softplusf(float x) {
    return (x > 20.0f) ? x : log1pf(expf(x));
}

// ---------------------------------------------------------------------------
// Zero-fill (emb region of d_out and the e workspace must start at 0 because
// the GEMM epilogue accumulates into them with atomics).
// ---------------------------------------------------------------------------
__global__ void zero_kernel(float* __restrict__ p, long long n) {
    long long i = (long long)blockIdx.x * blockDim.x + threadIdx.x;
    long long stride = (long long)gridDim.x * blockDim.x;
    for (; i < n; i += stride) p[i] = 0.0f;
}

// ---------------------------------------------------------------------------
// Fused GEMM: h = relu(X @ W1 + b1) for a 128-row x 128-col tile per block,
// computed with v_wmma_f32_16x16x32_bf16.  h is never written to memory;
// instead we fold it into:
//   emb[b][n]   += h[r][n]                      (batch-segment atomics)
//   e_ws[r][j]  += sum_n h[r][n] * W2[n][j]     (shuffle-reduce + atomics)
// ---------------------------------------------------------------------------
__global__ __launch_bounds__(256) void gemm_h_fused(
    const float* __restrict__ X,    // [ROWS, 128]
    const float* __restrict__ W1,   // [128, 128]
    const float* __restrict__ b1,   // [128]
    const float* __restrict__ W2,   // [128, 2]
    float* __restrict__ emb,        // [BSZ, 128]  (zeroed)
    float* __restrict__ e_ws)       // [ROWS, 2]   (zeroed)
{
    __shared__ unsigned int   ldsA[128 * 64];     // 128 rows x 128 K, bf16 pairs
    __shared__ unsigned short ldsW[128 * 128];    // W1 transposed: [n][k] bf16

    const int tid = threadIdx.x;
    const long long R0 = (long long)blockIdx.x * 128;

    // Stage A tile: 128x128 fp32 -> bf16 (float4 loads, packed u32 stores)
    #pragma unroll
    for (int i = 0; i < 16; ++i) {
        int e4  = tid + 256 * i;          // 0..4095 float4s
        int row = e4 >> 5;
        int c4  = e4 & 31;
        const float4 f = ((const float4*)(X + (R0 + row) * KDIM))[c4];
        ldsA[row * 64 + c4 * 2 + 0] =
            (unsigned)bf16rne(f.x) | ((unsigned)bf16rne(f.y) << 16);
        ldsA[row * 64 + c4 * 2 + 1] =
            (unsigned)bf16rne(f.z) | ((unsigned)bf16rne(f.w) << 16);
    }
    // Stage W1 transposed: ldsW[n*128 + k] = bf16(W1[k][n])
    #pragma unroll
    for (int i = 0; i < 64; ++i) {
        int e = tid + 256 * i;            // 0..16383
        int k = e >> 7, n = e & 127;
        ldsW[n * 128 + k] = bf16rne(W1[k * 128 + n]);
    }
    __syncthreads();

    const int w    = tid >> 5;
    const int lane = tid & 31;
    const int half = lane >> 4;
    const int l16  = lane & 15;
    const int wm   = w & 3;              // 4 waves along M (32 rows each)
    const int wn   = w >> 2;             // 2 waves along N (64 cols each)
    const int m0   = wm * 32;
    const int n0   = wn * 64;

    v8f zero;
    #pragma unroll
    for (int e = 0; e < 8; ++e) zero[e] = 0.0f;
    v8f acc[8];
    #pragma unroll
    for (int t = 0; t < 8; ++t) acc[t] = zero;

    union FragU { unsigned u[8]; uint4 q[2]; v16bf v; };

    #pragma unroll
    for (int kc = 0; kc < 4; ++kc) {
        const int k0 = kc * 32;

        // A fragments (16x32 bf16): lane l16 = row, paired-K across 8 VGPRs
        FragU af[2];
        #pragma unroll
        for (int mt = 0; mt < 2; ++mt) {
            int m = m0 + mt * 16 + l16;
            #pragma unroll
            for (int v = 0; v < 8; ++v) {
                int kp = 16 * half + 2 * v;
                af[mt].u[v] = ldsA[m * 64 + ((k0 + kp) >> 1)];
            }
        }
        // B fragments (32x16 bf16): lane l16 = col, 16 contiguous K per lane
        FragU bfr[4];
        #pragma unroll
        for (int nt = 0; nt < 4; ++nt) {
            int n = n0 + nt * 16 + l16;
            const uint4* src =
                (const uint4*)&ldsW[n * 128 + k0 + 16 * half]; // 32B aligned
            bfr[nt].q[0] = src[0];
            bfr[nt].q[1] = src[1];
        }
        #pragma unroll
        for (int mt = 0; mt < 2; ++mt)
            #pragma unroll
            for (int nt = 0; nt < 4; ++nt)
                acc[mt * 4 + nt] = __builtin_amdgcn_wmma_f32_16x16x32_bf16(
                    false, af[mt].v, false, bfr[nt].v,
                    (short)0, acc[mt * 4 + nt], false, false);
    }

    // Epilogue: bias + relu, then fused emb / e reductions.
    #pragma unroll
    for (int mt = 0; mt < 2; ++mt) {
        const long long rbase = R0 + m0 + mt * 16 + 8 * half; // rows rbase+v
        #pragma unroll
        for (int nt = 0; nt < 4; ++nt) {
            const int   n   = n0 + nt * 16 + l16;
            const float w2a = W2[n * 2 + 0];
            const float w2b = W2[n * 2 + 1];
            const float bb  = b1[n];

            float     embsum = 0.0f;
            long long curb   = rbase / NN;
            #pragma unroll
            for (int v = 0; v < 8; ++v) {
                const long long r  = rbase + v;
                float hv = acc[mt * 4 + nt][v] + bb;
                hv = hv > 0.0f ? hv : 0.0f;

                const long long bcur = r / NN;
                if (bcur != curb) {
                    atomicAdd(&emb[curb * HDIM + n], embsum);
                    embsum = 0.0f;
                    curb = bcur;
                }
                embsum += hv;

                // e[r][j] partials: reduce over 16 lanes (cols) of this half
                float v0 = hv * w2a;
                float v1 = hv * w2b;
                #pragma unroll
                for (int s = 8; s >= 1; s >>= 1) {
                    v0 += __shfl_xor(v0, s, 16);
                    v1 += __shfl_xor(v1, s, 16);
                }
                if (l16 == 0) {
                    atomicAdd(&e_ws[r * 2 + 0], v0);
                    atomicAdd(&e_ws[r * 2 + 1], v1);
                }
            }
            atomicAdd(&emb[curb * HDIM + n], embsum);
        }
    }
}

// ---------------------------------------------------------------------------
// e -> symmetrize, softplus, split into D_init (zero diagonal) and Wmat.
// ---------------------------------------------------------------------------
__global__ void finalize_e(const float* __restrict__ e_ws,
                           const float* __restrict__ b2,
                           float* __restrict__ Dout,
                           float* __restrict__ Wout)
{
    const int b = blockIdx.x;
    const int t = threadIdx.x;
    if (t >= NN) return;
    const int i = t / NMAX, j = t % NMAX;
    const long long base = (long long)b * NN;

    float a0 = e_ws[(base + t) * 2 + 0] + e_ws[(base + j * NMAX + i) * 2 + 0]
             + 2.0f * b2[0];
    float a1 = e_ws[(base + t) * 2 + 1] + e_ws[(base + j * NMAX + i) * 2 + 1]
             + 2.0f * b2[1];

    Dout[base + t] = (i == j) ? 0.0f : softplusf(a0);
    Wout[base + t] = softplusf(a1);
}

// ---------------------------------------------------------------------------
// dist_nlsq: one wave32 per batch element.  Gram matrix, 3-component deflated
// power iteration (PRNG approximated by PCG+Box-Muller, normalization
// approximated per-batch), then 100 steps of tanh-clipped gradient descent.
// ---------------------------------------------------------------------------
__global__ __launch_bounds__(32) void nlsq_kernel(
    const float* __restrict__ Dmat,
    const float* __restrict__ Wmat,
    float* __restrict__ Xout)
{
    __shared__ float G[NN];
    __shared__ float Dl[NN];
    __shared__ float Wl[NN];
    __shared__ float rowsum[NMAX];
    __shared__ float colsum[NMAX];
    __shared__ float U[NMAX];
    __shared__ float Xs[NMAX][3];

    const int b = blockIdx.x;
    const int t = threadIdx.x;
    const long long base = (long long)b * NN;

    for (int e = t; e < NN; e += 32) {
        float d = Dmat[base + e];
        Dl[e] = d;
        Wl[e] = Wmat[base + e];
        G[e]  = d * d;                       // Dsq
    }
    __syncthreads();

    if (t < NMAX) {
        float rs = 0.0f, cs = 0.0f;
        for (int j = 0; j < NMAX; ++j) {
            rs += G[t * NMAX + j];
            cs += G[j * NMAX + t];
        }
        rowsum[t] = rs / (float)NMAX;        // D_col[i]
        colsum[t] = cs / (float)NMAX;        // D_row[j]
    }
    __syncthreads();

    float mean = 0.0f;
    for (int i2 = 0; i2 < NMAX; ++i2) mean += rowsum[i2];
    mean /= (float)NMAX;                     // sum(Dsq)/N^2
    __syncthreads();

    for (int e = t; e < NN; e += 32) {
        int i = e / NMAX, j = e % NMAX;
        G[e] = -0.5f * (G[e] - colsum[j] - rowsum[i] + mean);
    }
    __syncthreads();

    // deflated power iteration, k = 3 components, 10 steps each
    for (int c = 0; c < 3; ++c) {
        if (t < NMAX)
            U[t] = gaussf((unsigned)(b * 1315423911u) ^ (unsigned)(c * 2654435761u)
                          ^ (unsigned)(t * 97u) ^ 0xA5A5u);
        __syncthreads();
        for (int it = 0; it < 10; ++it) {
            float nrm = 0.0f;
            for (int i2 = 0; i2 < NMAX; ++i2) nrm += U[i2] * U[i2];
            nrm = fmaxf(sqrtf(nrm), 1e-3f);
            float nu = 0.0f;
            if (t < NMAX)
                for (int j = 0; j < NMAX; ++j)
                    nu += G[t * NMAX + j] * (U[j] / nrm);
            __syncthreads();
            if (t < NMAX) U[t] = nu;
            __syncthreads();
        }
        float eig = 0.0f;
        for (int i2 = 0; i2 < NMAX; ++i2) eig += U[i2] * U[i2];
        float inv = 1.0f / sqrtf(sqrtf(eig + 0.01f));   // (eig+eps2)^-0.25
        __syncthreads();
        if (t < NMAX) {
            U[t] *= inv;
            Xs[t][c] = U[t];
        }
        __syncthreads();
        for (int e = t; e < NN; e += 32)
            G[e] -= U[e / NMAX] * U[e % NMAX];          // deflate
        __syncthreads();
    }

    // X0 += noise
    if (t < NMAX)
        for (int c = 0; c < 3; ++c)
            Xs[t][c] += gaussf((unsigned)(b * 2246822519u) ^ 0x3779B900u
                               ^ (unsigned)(t * 193u) ^ (unsigned)(c * 3266489917u));
    __syncthreads();

    // 100-step gradient descent with tanh speed clipping
    for (int step = 0; step < 100; ++step) {
        float nx0 = 0.0f, nx1 = 0.0f, nx2 = 0.0f;
        if (t < NMAX) {
            const float xi0 = Xs[t][0], xi1 = Xs[t][1], xi2 = Xs[t][2];
            float g0 = 0.0f, g1 = 0.0f, g2 = 0.0f;
            for (int j = 0; j < NMAX; ++j) {
                float d0 = xi0 - Xs[j][0];
                float d1 = xi1 - Xs[j][1];
                float d2 = xi2 - Xs[j][2];
                float dx = d0 * d0 + d1 * d1 + d2 * d2 + 0.01f;
                float ws = Wl[t * NMAX + j] + Wl[j * NMAX + t];
                float cf = -4.0f * ws * (Dl[t * NMAX + j] - dx);
                g0 += cf * d0; g1 += cf * d1; g2 += cf * d2;
            }
            float dX0 = -0.1f * g0, dX1 = -0.1f * g1, dX2 = -0.1f * g2;
            float speed = sqrtf(dX0 * dX0 + dX1 * dX1 + dX2 * dX2 + 1e-3f);
            float alpha_t = 0.1f + 4.9f * ((100.0f - (float)step) / 100.0f);
            float scale = alpha_t * tanhf(speed / alpha_t) / speed;
            nx0 = xi0 + dX0 * scale;
            nx1 = xi1 + dX1 * scale;
            nx2 = xi2 + dX2 * scale;
        }
        __syncthreads();
        if (t < NMAX) { Xs[t][0] = nx0; Xs[t][1] = nx1; Xs[t][2] = nx2; }
        __syncthreads();
    }

    if (t < NMAX) {
        Xout[(long long)b * (NMAX * 3) + t * 3 + 0] = Xs[t][0];
        Xout[(long long)b * (NMAX * 3) + t * 3 + 1] = Xs[t][1];
        Xout[(long long)b * (NMAX * 3) + t * 3 + 2] = Xs[t][2];
    }
}

// ---------------------------------------------------------------------------
extern "C" void kernel_launch(void* const* d_in, const int* in_sizes, int n_in,
                              void* d_out, int out_size, void* d_ws, size_t ws_size,
                              hipStream_t stream) {
    (void)in_sizes; (void)n_in; (void)out_size; (void)ws_size;

    const float* X  = (const float*)d_in[0];   // [ROWS, 128]
    const float* W1 = (const float*)d_in[1];   // [128, 128]
    const float* b1 = (const float*)d_in[2];   // [128]
    const float* W2 = (const float*)d_in[3];   // [128, 2]
    const float* b2 = (const float*)d_in[4];   // [2]
    // d_in[5] = batch_size (2048, compile-time constant here)

    float* out  = (float*)d_out;
    float* Dout = out;                               // [BSZ, 21, 21]
    float* Wout = out + (long long)ROWS;             // [BSZ, 21, 21]
    float* emb  = out + 2LL * ROWS;                  // [BSZ, 128]
    float* Xout = out + 2LL * ROWS + (long long)BSZ * HDIM; // [BSZ, 21, 3]

    float* e_ws = (float*)d_ws;                      // [ROWS, 2] partial e

    zero_kernel<<<1024, 256, 0, stream>>>(emb,  (long long)BSZ * HDIM);
    zero_kernel<<<2048, 256, 0, stream>>>(e_ws, (long long)ROWS * 2);

    gemm_h_fused<<<ROWS / 128, 256, 0, stream>>>(X, W1, b1, W2, emb, e_ws);

    finalize_e<<<BSZ, 448, 0, stream>>>(e_ws, b2, Dout, Wout);

    nlsq_kernel<<<BSZ, 32, 0, stream>>>(Dout, Wout, Xout);
}
